// Encoder_52072183497085
// MI455X (gfx1250) — matrix operand order repaired and checked by reading, not compiled, not run on hardware
//
#include <hip/hip_runtime.h>
#include <hip/hip_bf16.h>
#include <math.h>

// ---------------- types / helpers ----------------
typedef __attribute__((ext_vector_type(16))) __bf16 bf16x16;
typedef __attribute__((ext_vector_type(8)))  float  f32x8;

static __device__ __forceinline__ unsigned short f2bf(float f) {
    union { float f; unsigned u; } x; x.f = f;
    unsigned r = (x.u + 0x7FFFu + ((x.u >> 16) & 1u)) >> 16;   // RNE
    return (unsigned short)r;
}
static __device__ __forceinline__ float bf2f(unsigned short h) {
    union { unsigned u; float f; } x; x.u = ((unsigned)h) << 16;
    return x.f;
}
static __device__ __forceinline__ float sigm(float x) {
    return 1.0f / (1.0f + __expf(-x));
}
static __device__ __forceinline__ f32x8 wmma_bf16(bf16x16 a, bf16x16 b, f32x8 c) {
    return __builtin_amdgcn_wmma_f32_16x16x32_bf16(false, a, false, b, (short)0, c,
                                                   false, false);
}
// async global->LDS 16B copy (per lane), tracked by ASYNCcnt
static __device__ __forceinline__ void async_copy16(unsigned lds_addr,
                                                    const void* gaddr) {
    asm volatile("global_load_async_to_lds_b128 %0, %1, off"
                 :: "v"(lds_addr), "v"((unsigned long long)gaddr) : "memory");
}
static __device__ __forceinline__ void wait_async0() {
    asm volatile("s_wait_asynccnt 0x0" ::: "memory");
}

// -------- weight prep: [EC,Cin,3,3] f32 -> [EC,9,Cin] bf16 (tap-major) -----
__global__ __launch_bounds__(256) void wprep_kernel(
    const float* __restrict__ Wf,
    unsigned short* __restrict__ Wbf,
    int Cin)
{
    const size_t oc = blockIdx.x;      // E*Cout
    const int tap   = blockIdx.y;      // 0..8
    for (int ic = threadIdx.x; ic < Cin; ic += 256)
        Wbf[(oc * 9 + tap) * Cin + ic] = f2bf(Wf[(oc * Cin + ic) * 9 + tap]);
}

// ---------------- conv0: 1x1 conv 3->32 + tanh, bf16 out ----------------
__global__ __launch_bounds__(256) void conv0_kernel(
    const float* __restrict__ x,      // [64,3,64,64]
    const float* __restrict__ w0,     // [E,32,3]
    const float* __restrict__ b0,     // [E,32]
    const int* __restrict__ label,
    unsigned short* __restrict__ out) // [64,32,4096] bf16
{
    const int b = blockIdx.z;
    const int p = blockIdx.x * 256 + threadIdx.x;   // 0..4095
    const int e = label[b];
    const float x0 = x[((size_t)b * 3 + 0) * 4096 + p];
    const float x1 = x[((size_t)b * 3 + 1) * 4096 + p];
    const float x2 = x[((size_t)b * 3 + 2) * 4096 + p];
    const float* w  = w0 + (size_t)e * 96;
    const float* bb = b0 + (size_t)e * 32;
    #pragma unroll 4
    for (int oc = 0; oc < 32; ++oc) {
        float v = tanhf(w[oc * 3 + 0] * x0 + w[oc * 3 + 1] * x1 +
                        w[oc * 3 + 2] * x2 + bb[oc]);
        out[((size_t)b * 32 + oc) * 4096 + p] = f2bf(v);
    }
}

// ---------------- implicit-GEMM 3x3 conv via WMMA bf16 ----------------
// Tile M=128 x N=64, K=32/step, tap-major K. 8 waves x (2x2) accumulators
// -> 4 v_wmma per K-step. Double-buffered LDS; weights DMA'd with
// global_load_async_to_lds_b128 and overlapped with the WMMAs.
#define CMT 128
#define CNT 64
#define LDA 48   // padded LDS row (bf16 elems) -> 96B, 16B-aligned rows

__global__ __launch_bounds__(256) void conv3x3_wmma_kernel(
    const unsigned short* __restrict__ X,    // bf16 [B,Cin,Hin,Win]
    const unsigned short* __restrict__ Wbf,  // bf16 [E,Cout,9,Cin] tap-major
    const float* __restrict__ Bs,            // f32  [E,Cout]
    float* __restrict__ Y,                   // f32  [B,Cout,Hout,Wout]
    const int* __restrict__ label,
    int Cin, int Cout, int Hin, int Win, int Hout, int Wout, int stride,
    int cbShift)                             // log2(Cin/32)
{
    __shared__ alignas(16) unsigned short lA[2][CMT * LDA];
    __shared__ alignas(16) unsigned short lB[2][CNT * LDA];

    const int tid  = threadIdx.x;
    const int b    = blockIdx.z;
    const int e    = label[b];
    const int S    = Hout * Wout;
    const int HWin = Hin * Win;
    const int ocBase = blockIdx.x * CMT;
    const int nBase  = blockIdx.y * CNT;
    const int cb     = 1 << cbShift;
    const int nsteps = 9 << cbShift;

    const unsigned short* Wb = Wbf + ((size_t)e * Cout + ocBase) * 9 * Cin;
    const unsigned short* Xb = X + (size_t)b * Cin * HWin;

    const int wave = tid >> 5, lane = tid & 31;
    const int msub = wave >> 1, nsub = wave & 1;
    const int g = lane >> 4, lr = lane & 15;

    // B-fill mapping: fixed pixel per thread, 8 contiguous kk per thread
    const int fn  = tid & 63;            // pixel in tile
    const int fk0 = (tid >> 6) * 8;      // kk base: 0,8,16,24
    const int p   = nBase + fn;
    const int ho  = p / Wout, wo = p - ho * Wout;

    f32x8 acc[2][2];
    #pragma unroll
    for (int i = 0; i < 2; ++i)
        #pragma unroll
        for (int j = 0; j < 2; ++j)
            #pragma unroll
            for (int r = 0; r < 8; ++r) acc[i][j][r] = 0.0f;

    auto fill = [&](int step, int sel) {
        const int tap = step >> cbShift;
        const int ic0 = (step & (cb - 1)) << 5;
        // --- A tile: 128x32 bf16 via async DMA (2 x 16B per thread) ---
        const unsigned short* Wtap = Wb + (size_t)tap * Cin + ic0;
        #pragma unroll
        for (int t = 0; t < 2; ++t) {
            int idx = tid + t * 256;              // 0..511
            int m = idx >> 2, part = idx & 3;
            unsigned dst = (unsigned)(unsigned long long)
                           &lA[sel][m * LDA + part * 8];
            async_copy16(dst, &Wtap[(size_t)m * 9 * Cin + part * 8]);
        }
        // --- B tile: branchless clamped gather, one ds_store_b128/thread ---
        const int kh = tap / 3, kw = tap - kh * 3;
        const int hi = ho * stride + kh - 1;
        const int wi = wo * stride + kw - 1;
        const bool inb = (hi >= 0) & (hi < Hin) & (wi >= 0) & (wi < Win);
        const int hic = min(max(hi, 0), Hin - 1);
        const int wic = min(max(wi, 0), Win - 1);
        const unsigned short* bb =
            Xb + (size_t)hic * Win + wic + (size_t)(ic0 + fk0) * HWin;
        union { unsigned short s[8]; uint4 q; } v;
        #pragma unroll
        for (int t = 0; t < 8; ++t) {
            unsigned short xv = bb[(size_t)t * HWin];
            v.s[t] = inb ? xv : (unsigned short)0;
        }
        *(uint4*)&lB[sel][fn * LDA + fk0] = v.q;
    };

    fill(0, 0);
    wait_async0();
    __syncthreads();

    for (int step = 0; step < nsteps; ++step) {
        const int sel = step & 1;
        if (step + 1 < nsteps) fill(step + 1, sel ^ 1);

        // --- fragments (ISA 16-bit A/B lane layouts) ---
        union { uint4 q[2]; bf16x16 v; } a0, a1, b0, b1;
        const unsigned short* ar0 = &lA[sel][(msub * 32 + lr) * LDA];
        const unsigned short* ar1 = ar0 + 16 * LDA;
        a0.q[0] = *(const uint4*)(ar0 + 8 * g);
        a0.q[1] = *(const uint4*)(ar0 + 16 + 8 * g);
        a1.q[0] = *(const uint4*)(ar1 + 8 * g);
        a1.q[1] = *(const uint4*)(ar1 + 16 + 8 * g);
        const unsigned short* br0 = &lB[sel][(nsub * 32 + lr) * LDA];
        const unsigned short* br1 = br0 + 16 * LDA;
        b0.q[0] = *(const uint4*)(br0 + 8 * g);
        b0.q[1] = *(const uint4*)(br0 + 16 + 8 * g);
        b1.q[0] = *(const uint4*)(br1 + 8 * g);
        b1.q[1] = *(const uint4*)(br1 + 16 + 8 * g);

        acc[0][0] = wmma_bf16(a0.v, b0.v, acc[0][0]);
        acc[0][1] = wmma_bf16(a0.v, b1.v, acc[0][1]);
        acc[1][0] = wmma_bf16(a1.v, b0.v, acc[1][0]);
        acc[1][1] = wmma_bf16(a1.v, b1.v, acc[1][1]);

        wait_async0();       // async A-DMA for next step done (overlapped with WMMA)
        __syncthreads();
    }

    // --- epilogue: bias add, store f32 (C/D layout: elem r -> M=r+8g, N=lr) ---
    #pragma unroll
    for (int ms = 0; ms < 2; ++ms) {
        #pragma unroll
        for (int ns = 0; ns < 2; ++ns) {
            const int n = nBase + nsub * 32 + ns * 16 + lr;
            #pragma unroll
            for (int r = 0; r < 8; ++r) {
                int oc = ocBase + msub * 32 + ms * 16 + r + 8 * g;
                Y[((size_t)b * Cout + oc) * S + n] =
                    acc[ms][ns][r] + Bs[(size_t)e * Cout + oc];
            }
        }
    }
}

// ---------------- ConvLSTM gate math (zero initial state) ----------------
__global__ __launch_bounds__(256) void lstm_gate_kernel(
    const float* __restrict__ gates,   // [B,4co,S]
    unsigned short* __restrict__ h,    // [B,co,S] bf16
    int sShift, int coShift, size_t total)
{
    size_t i = (size_t)blockIdx.x * 256 + threadIdx.x;
    if (i >= total) return;
    const int S = 1 << sShift, co = 1 << coShift;
    int s = (int)(i & (S - 1));
    size_t t = i >> sShift;
    int c = (int)(t & (co - 1));
    int b = (int)(t >> coShift);
    const float* gb = gates + ((size_t)b << (2 + coShift + sShift));
    float gi = gb[((size_t)c << sShift) + s];
    float gg = gb[(((size_t)(2 * co) + c) << sShift) + s];
    float go = gb[(((size_t)(3 * co) + c) << sShift) + s];
    float cc = sigm(gi) * tanhf(gg);
    h[i] = f2bf(sigm(go) * tanhf(cc));
}

// ---------------- gate math + stride-2 shortcut residual ----------------
__global__ __launch_bounds__(256) void lstm_gate_res_kernel(
    const float* __restrict__ gates,   // [B,4co,S]
    const float* __restrict__ sc,      // [B,co,S]
    unsigned short* __restrict__ xout, // [B,co,S] bf16
    int sShift, int coShift, size_t total)
{
    size_t i = (size_t)blockIdx.x * 256 + threadIdx.x;
    if (i >= total) return;
    const int S = 1 << sShift, co = 1 << coShift;
    int s = (int)(i & (S - 1));
    size_t t = i >> sShift;
    int c = (int)(t & (co - 1));
    int b = (int)(t >> coShift);
    const float* gb = gates + ((size_t)b << (2 + coShift + sShift));
    float gi = gb[((size_t)c << sShift) + s];
    float gg = gb[(((size_t)(2 * co) + c) << sShift) + s];
    float go = gb[(((size_t)(3 * co) + c) << sShift) + s];
    float cc = sigm(gi) * tanhf(gg);
    float hv = sigm(go) * tanhf(cc);
    xout[i] = f2bf(hv + sc[i]);
}

// ---------------- convout: 1x1 conv 512->32 + tanh, to d_out ----------------
__global__ __launch_bounds__(256) void convout_kernel(
    const unsigned short* __restrict__ x3,   // bf16 [64,512,64]
    const float* __restrict__ wout,          // [E,32,512]
    const float* __restrict__ bout,          // [E,32]
    const int* __restrict__ label,
    float* __restrict__ out)                 // [64,32,64]
{
    int idx = blockIdx.x * 256 + threadIdx.x;     // (b*32+oc)*64+s
    int s  = idx & 63;
    int t  = idx >> 6;
    int oc = t & 31;
    int b  = t >> 5;
    int e  = label[b];
    const float* w = wout + ((size_t)e * 32 + oc) * 512;
    const unsigned short* xb = x3 + (size_t)b * 512 * 64;
    float acc = bout[(size_t)e * 32 + oc];
    #pragma unroll 8
    for (int ic = 0; ic < 512; ++ic)
        acc += w[ic] * bf2f(xb[(size_t)ic * 64 + s]);
    out[idx] = tanhf(acc);
}

// ---------------- host launch ----------------
static int ilog2i(int v) { int r = 0; while (v > 1) { v >>= 1; ++r; } return r; }

extern "C" void kernel_launch(void* const* d_in, const int* in_sizes, int n_in,
                              void* d_out, int out_size, void* d_ws, size_t ws_size,
                              hipStream_t stream) {
    (void)in_sizes; (void)n_in; (void)out_size; (void)ws_size;
    const float* x     = (const float*)d_in[0];
    const int*   label = (const int*)d_in[1];
    const float* w0    = (const float*)d_in[2];
    const float* b0    = (const float*)d_in[3];
    const float* wout  = (const float*)d_in[22];
    const float* bout  = (const float*)d_in[23];

    struct StageW { const float *win0, *bin0, *win1, *bin1, *wsc, *bsc; };
    StageW sw[3];
    for (int s = 0; s < 3; ++s) {
        int base = 4 + s * 6;
        sw[s].win0 = (const float*)d_in[base + 0];
        sw[s].bin0 = (const float*)d_in[base + 1];
        sw[s].win1 = (const float*)d_in[base + 2];
        sw[s].bin1 = (const float*)d_in[base + 3];
        sw[s].wsc  = (const float*)d_in[base + 4];
        sw[s].bsc  = (const float*)d_in[base + 5];
    }

    const int B = 64, E = 4;
    char* ws = (char*)d_ws;
    size_t off = 0;
    auto carve = [&](size_t bytes) -> void* {
        void* p = ws + off;
        off += (bytes + 255) & ~(size_t)255;
        return p;
    };
    float* gates = (float*)carve((size_t)B * 512 * 1024 * sizeof(float));      // 134 MB
    float* scbuf = (float*)carve((size_t)B * 128 * 1024 * sizeof(float));      // 33.5 MB
    unsigned short* actA = (unsigned short*)carve((size_t)B * 32 * 4096 * 2);  // 16.8 MB
    unsigned short* actB = (unsigned short*)carve((size_t)B * 32 * 4096 * 2);  // 16.8 MB
    unsigned short* h1b  = (unsigned short*)carve((size_t)B * 32 * 4096 * 2);  // 16.8 MB
    unsigned short* wbf  = (unsigned short*)carve((size_t)E * 2048 * 512 * 9 * 2); // 75.5 MB

    auto prep = [&](const float* wsrc, int Cout, int Cin) {
        wprep_kernel<<<dim3((unsigned)(E * Cout), 9), 256, 0, stream>>>(wsrc, wbf, Cin);
    };

    // stem conv
    conv0_kernel<<<dim3(16, 1, B), 256, 0, stream>>>(x, w0, b0, label, actA);

    const int ci_arr[3]  = {32, 128, 256};
    const int co_arr[3]  = {128, 256, 512};
    const int hin_arr[3] = {64, 32, 16};

    unsigned short* cur = actA;
    unsigned short* nxt = actB;
    for (int s = 0; s < 3; ++s) {
        const int ci = ci_arr[s], co = co_arr[s];
        const int Hin = hin_arr[s], Hout = Hin >> 1;
        const int S = Hout * Hout;
        const size_t totalH = (size_t)B * co * S;
        const unsigned blocksH = (unsigned)((totalH + 255) / 256);
        const int sSh = ilog2i(S), coSh = ilog2i(co);
        const int cbCi = ilog2i(ci >> 5), cbCo = ilog2i(co >> 5);

        // gates = conv3x3(cur, win0, stride 2) : ci -> 4co
        prep(sw[s].win0, 4 * co, ci);
        conv3x3_wmma_kernel<<<dim3(4 * co / CMT, S / CNT, B), 256, 0, stream>>>(
            cur, wbf, sw[s].bin0, gates, label,
            ci, 4 * co, Hin, Hin, Hout, Hout, 2, cbCi);
        // h1 = lstm_step(gates)
        lstm_gate_kernel<<<blocksH, 256, 0, stream>>>(gates, h1b, sSh, coSh, totalH);
        // gates = conv3x3(h1, win1, stride 1) : co -> 4co
        prep(sw[s].win1, 4 * co, co);
        conv3x3_wmma_kernel<<<dim3(4 * co / CMT, S / CNT, B), 256, 0, stream>>>(
            h1b, wbf, sw[s].bin1, gates, label,
            co, 4 * co, Hout, Hout, Hout, Hout, 1, cbCo);
        // sc = conv3x3(cur, wsc, stride 2) : ci -> co
        prep(sw[s].wsc, co, ci);
        conv3x3_wmma_kernel<<<dim3(co / CMT, S / CNT, B), 256, 0, stream>>>(
            cur, wbf, sw[s].bsc, scbuf, label,
            ci, co, Hin, Hin, Hout, Hout, 2, cbCi);
        // x_next = lstm_step(gates) + sc
        lstm_gate_res_kernel<<<blocksH, 256, 0, stream>>>(gates, scbuf, nxt,
                                                          sSh, coSh, totalH);

        unsigned short* tmp = cur; cur = nxt; nxt = tmp;
    }

    // head conv 512->32 + tanh, scatter to d_out
    convout_kernel<<<dim3((B * 32 * 64) / 256), 256, 0, stream>>>(
        cur, wout, bout, label, (float*)d_out);
}
